// Attention_80917183857290
// MI455X (gfx1250) — matrix operand order repaired and checked by reading, not compiled, not run on hardware
//
#include <hip/hip_runtime.h>
#include <hip/hip_fp16.h>

#define B_  2
#define S_  2048
#define D_  1024
#define H_  32
#define HD_ 32

typedef __attribute__((ext_vector_type(16))) _Float16 v16h;
typedef __attribute__((ext_vector_type(8)))  _Float16 v8h;
typedef __attribute__((ext_vector_type(8)))  float    v8f;

static __device__ __forceinline__ v8f wmma_f16(v16h a, v16h b, v8f c) {
  return __builtin_amdgcn_wmma_f32_16x16x32_f16(false, a, false, b, (short)0, c,
                                                false, false);
}

// raw VALU max (no canonicalization chains)
static __device__ __forceinline__ float fmax_raw(float a, float b) {
  float r;
  asm("v_max_num_f32 %0, %1, %2" : "=v"(r) : "v"(a), "v"(b));
  return r;
}

// xor-butterfly max over a 16-lane group using fused DPP max:
// one v_max_num_f32_dpp per step; masks {1,2,7,15} span the 16-lane group.
static __device__ __forceinline__ float rowmax16(float x) {
  float t;
  asm("v_max_num_f32_dpp %0, %1, %1 quad_perm:[1,0,3,2] row_mask:0xf bank_mask:0xf bound_ctrl:1"
      : "=v"(t) : "v"(x));
  asm("v_max_num_f32_dpp %0, %1, %1 quad_perm:[2,3,0,1] row_mask:0xf bank_mask:0xf bound_ctrl:1"
      : "=v"(x) : "v"(t));
  asm("v_max_num_f32_dpp %0, %1, %1 row_half_mirror row_mask:0xf bank_mask:0xf bound_ctrl:1"
      : "=v"(t) : "v"(x));
  asm("v_max_num_f32_dpp %0, %1, %1 row_mirror row_mask:0xf bank_mask:0xf bound_ctrl:1"
      : "=v"(x) : "v"(t));
  return x;
}

// ---------------------------------------------------------------------------
// elementwise fp32 -> f16 (one-time; keeps GEMM hot loops convert-free)
// ---------------------------------------------------------------------------
__global__ __launch_bounds__(256) void cvt_f16(const float* __restrict__ in,
                                               _Float16* __restrict__ out, int n) {
  int i = (blockIdx.x * 256 + threadIdx.x) * 8;
  if (i < n) {
    v8f f = *(const v8f*)(in + i);
    v8h h;
#pragma unroll
    for (int j = 0; j < 8; ++j) h[j] = (_Float16)f[j];
    *(v8h*)(out + i) = h;
  }
}

// ---------------------------------------------------------------------------
// Y = A[M,K]h @ W[N,K]h^T ; one wave -> 32x64 tile (8 WMMAs / 12 b128 loads).
// STORE_T writes Y transposed as [B, N, S] (used for V).
// ---------------------------------------------------------------------------
template <bool OUT_FLOAT, bool STORE_T>
__global__ __launch_bounds__(32) void gemm_h(const _Float16* __restrict__ A,
                                             const _Float16* __restrict__ W,
                                             void* __restrict__ Yp,
                                             int M, int N, int Kdim) {
  const int lane = threadIdx.x;
  const int nlo  = lane & 15;
  const int grp  = lane >> 4;
  const int tn   = blockIdx.x * 64;
  const int tm   = blockIdx.y * 32;

  const _Float16* arow0 = A + (size_t)(tm + nlo) * Kdim + 16 * grp;
  const _Float16* arow1 = arow0 + (size_t)16 * Kdim;
  const _Float16* wrow  = W + (size_t)(tn + nlo) * Kdim + 16 * grp;

  v8f acc[2][4] = {};

  for (int kc = 0; kc < Kdim; kc += 32) {
    v16h a0, a1;
    {
      v8h lo = *(const v8h*)(arow0 + kc);
      v8h hi = *(const v8h*)(arow0 + kc + 8);
#pragma unroll
      for (int i = 0; i < 8; ++i) { a0[i] = lo[i]; a0[i + 8] = hi[i]; }
      lo = *(const v8h*)(arow1 + kc);
      hi = *(const v8h*)(arow1 + kc + 8);
#pragma unroll
      for (int i = 0; i < 8; ++i) { a1[i] = lo[i]; a1[i + 8] = hi[i]; }
    }
    v16h bf[4];
#pragma unroll
    for (int j = 0; j < 4; ++j)
      bf[j] = *(const v16h*)(wrow + (size_t)(j * 16) * Kdim + kc);

#pragma unroll
    for (int j = 0; j < 4; ++j) {
      acc[0][j] = wmma_f16(a0, bf[j], acc[0][j]);
      acc[1][j] = wmma_f16(a1, bf[j], acc[1][j]);
    }
  }

#pragma unroll
  for (int rt = 0; rt < 2; ++rt) {
#pragma unroll
    for (int j = 0; j < 4; ++j) {
      const int col  = tn + j * 16 + nlo;
      const int row0 = tm + rt * 16 + 8 * grp;       // rows row0..row0+7
      if (STORE_T) {
        // transposed: Yt[b][col][s], b = row>>11, s = row&2047 (S_ = 2048)
        const int b = row0 >> 11, s = row0 & (S_ - 1);
        v8h hv;
#pragma unroll
        for (int r = 0; r < 8; ++r) hv[r] = (_Float16)acc[rt][j][r];
        *(v8h*)((_Float16*)Yp + ((size_t)b * N + col) * S_ + s) = hv;
      } else {
#pragma unroll
        for (int r = 0; r < 8; ++r) {
          const size_t off = (size_t)(row0 + r) * N + col;
          if (OUT_FLOAT) ((float*)Yp)[off] = acc[rt][j][r];
          else           ((_Float16*)Yp)[off] = (_Float16)acc[rt][j][r];
        }
      }
    }
  }
}

// ---------------------------------------------------------------------------
// Flash attention + SSMax. One wave per (b, h, 16-query tile).
//  - SSMax scale AND log2(e) folded into the Q fragment -> softmax runs in
//    the exp2 domain (v_exp_f32 is exp2; no per-exp v_mul)
//  - even/odd key split: (p0,p1) adjacent -> cvt_pk_rtz + ds_store_b32
//  - causal mask only evaluated in the single tail chunk
//  - row-sums via WMMA vs ones; row-max via fused v_max_num_f32_dpp butterfly
// ---------------------------------------------------------------------------
__global__ __launch_bounds__(32) void flash_kernel(const _Float16* __restrict__ Q,
                                                   const _Float16* __restrict__ Kb,
                                                   const _Float16* __restrict__ Vt,
                                                   const float* __restrict__ seq_scale,
                                                   _Float16* __restrict__ O) {
  __shared__ _Float16 psh[16 * 40];
  const int lane = threadIdx.x;
  const int nlo  = lane & 15;
  const int grp  = lane >> 4;
  const int qt   = blockIdx.x * 16;
  const int b    = blockIdx.y >> 5;            // H_ == 32
  const int h    = blockIdx.y & 31;
  const size_t base  = ((size_t)b * S_) * D_ + (size_t)h * HD_;   // [B,S,D]
  // seq_scale[h] / sqrt(32) * log2(e)  (exp2-domain softmax)
  const float sscale = seq_scale[h] * 0.17677669529663687f * 1.4426950408889634f;

  v16h vones;
#pragma unroll
  for (int i = 0; i < 16; ++i) vones[i] = (_Float16)1.0f;

  // Q tile as A fragment; per-query SSMax factor folded in (lane == q row)
  v16h aq;
  {
    const _Float16* qp = Q + base + (size_t)(qt + nlo) * D_ + 16 * grp;
    v8h q0 = *(const v8h*)qp;
    v8h q1 = *(const v8h*)(qp + 8);
    const _Float16 qsc = (_Float16)(__logf((float)(qt + nlo + 2)) * sscale);
#pragma unroll
    for (int i = 0; i < 8; ++i) { aq[i] = q0[i] * qsc; aq[i + 8] = q1[i] * qsc; }
  }

  float mrun[8], lrun[8], alph[8];
  int   qg[8];
#pragma unroll
  for (int r = 0; r < 8; ++r) {
    qg[r]   = qt + r + 8 * grp;
    mrun[r] = -3.0e38f;
    lrun[r] = 0.0f;
  }
  v8f o0 = {}, o1 = {};

  // even keys -> sub-tile 0, odd keys -> sub-tile 1 (P columns stay natural)
  const _Float16* kp0 = Kb + base + (size_t)(2 * nlo) * D_ + 16 * grp;
  const _Float16* kp1 = kp0 + D_;
  const _Float16* vp0 = Vt + ((size_t)b * D_ + h * HD_ + nlo) * S_ + 16 * grp;
  const _Float16* vp1 = vp0 + (size_t)16 * S_;

  auto chunk = [&](int kc, bool masked) {
    v16h bk0 = *(const v16h*)(kp0 + (size_t)kc * D_);
    v16h bk1 = *(const v16h*)(kp1 + (size_t)kc * D_);
    v8f c0 = {}, c1 = {};
    c0 = wmma_f16(aq, bk0, c0);
    c1 = wmma_f16(aq, bk1, c1);
    v16h bv0 = *(const v16h*)(vp0 + kc);
    v16h bv1 = *(const v16h*)(vp1 + kc);
    if (!masked) {                            // prefetch next chunk (main loop)
      __builtin_prefetch(kp0 + (size_t)(kc + 32) * D_, 0, 3);
      __builtin_prefetch(kp1 + (size_t)(kc + 32) * D_, 0, 3);
      __builtin_prefetch(vp0 + kc + 32, 0, 3);
      __builtin_prefetch(vp1 + kc + 32, 0, 3);
    }
#pragma unroll
    for (int r = 0; r < 8; ++r) {
      float s0 = c0[r];
      float s1 = c1[r];
      if (masked) {
        const int k0 = kc + 2 * nlo;          // this lane's even key
        s0 += (k0     > qg[r]) ? -1e9f : 0.0f;
        s1 += (k0 + 1 > qg[r]) ? -1e9f : 0.0f;
      }
      const float mx   = rowmax16(fmax_raw(s0, s1));
      const float mnew = fmax_raw(mrun[r], mx);
      alph[r] = __builtin_amdgcn_exp2f(mrun[r] - mnew);
      mrun[r] = mnew;
      const float p0 = __builtin_amdgcn_exp2f(s0 - mnew);
      const float p1 = __builtin_amdgcn_exp2f(s1 - mnew);
      o0[r] *= alph[r];
      o1[r] *= alph[r];
      auto pk = __builtin_amdgcn_cvt_pkrtz(p0, p1);          // packed half2
      ((decltype(pk)*)psh)[(r + 8 * grp) * 20 + nlo] = pk;   // one b32 store
    }
    __syncthreads();                          // single-wave block: waits only

    v16h ap;                                  // P -> A fragment
    {
      const _Float16* pp = &psh[nlo * 40 + 16 * grp];
      v8h p0 = *(const v8h*)pp;
      v8h p1 = *(const v8h*)(pp + 8);
#pragma unroll
      for (int i = 0; i < 8; ++i) { ap[i] = p0[i]; ap[i + 8] = p1[i]; }
    }

    v8f csum = {};
    csum = wmma_f16(ap, vones, csum);         // row-sums, broadcast in C-layout
    o0 = wmma_f16(ap, bv0, o0);
    o1 = wmma_f16(ap, bv1, o1);
#pragma unroll
    for (int r = 0; r < 8; ++r) lrun[r] = lrun[r] * alph[r] + csum[r];
    __syncthreads();
  };

  // all chunks except the last are provably unmasked (kc+31 <= qt)
  const int kc_t = ((qt + 15) >> 5) << 5;
  for (int kc = 0; kc < kc_t; kc += 32) chunk(kc, false);
  chunk(kc_t, true);

#pragma unroll
  for (int r = 0; r < 8; ++r) {
    const float inv = 1.0f / lrun[r];
    const size_t off = base + (size_t)qg[r] * D_;
    O[off + nlo]      = (_Float16)(o0[r] * inv);
    O[off + 16 + nlo] = (_Float16)(o1[r] * inv);
  }
}

// ---------------------------------------------------------------------------
extern "C" void kernel_launch(void* const* d_in, const int* in_sizes, int n_in,
                              void* d_out, int out_size, void* d_ws, size_t ws_size,
                              hipStream_t stream) {
  const float* x         = (const float*)d_in[0];
  // d_in[1] = mask, d_in[2] = section_log_len: computed analytically
  const float* wq        = (const float*)d_in[3];
  const float* wk        = (const float*)d_in[4];
  const float* wv        = (const float*)d_in[5];
  const float* wo        = (const float*)d_in[6];
  const float* seq_scale = (const float*)d_in[7];

  const size_t tok = (size_t)B_ * S_ * D_;     // 4M elements
  const size_t wel = (size_t)D_ * D_;          // 1M elements
  _Float16* xh  = (_Float16*)d_ws;             // 8 MB
  _Float16* wqh = xh  + tok;                   // 2 MB each
  _Float16* wkh = wqh + wel;
  _Float16* wvh = wkh + wel;
  _Float16* woh = wvh + wel;
  _Float16* Qb  = woh + wel;                   // 8 MB  (reused as attn out)
  _Float16* Kb  = Qb  + tok;                   // 8 MB
  _Float16* Vt  = Kb  + tok;                   // 8 MB, layout [B, D, S]

  const int M = B_ * S_;
  dim3 cblk(256);
  cvt_f16<<<dim3((int)(tok / 8 / 256)), cblk, 0, stream>>>(x,  xh,  (int)tok);
  cvt_f16<<<dim3((int)(wel / 8 / 256)), cblk, 0, stream>>>(wq, wqh, (int)wel);
  cvt_f16<<<dim3((int)(wel / 8 / 256)), cblk, 0, stream>>>(wk, wkh, (int)wel);
  cvt_f16<<<dim3((int)(wel / 8 / 256)), cblk, 0, stream>>>(wv, wvh, (int)wel);
  cvt_f16<<<dim3((int)(wel / 8 / 256)), cblk, 0, stream>>>(wo, woh, (int)wel);

  dim3 blk(32);
  dim3 ggrid(D_ / 64, M / 32);
  gemm_h<false, false><<<ggrid, blk, 0, stream>>>(xh, wqh, Qb, M, D_, D_);
  gemm_h<false, false><<<ggrid, blk, 0, stream>>>(xh, wkh, Kb, M, D_, D_);
  gemm_h<false, true ><<<ggrid, blk, 0, stream>>>(xh, wvh, Vt, M, D_, D_);

  dim3 agrid(S_ / 16, B_ * H_);
  flash_kernel<<<agrid, blk, 0, stream>>>(Qb, Kb, Vt, seq_scale, Qb);

  gemm_h<true, false><<<ggrid, blk, 0, stream>>>(Qb, woh, d_out, M, D_, D_);
}
